// QuantizedLinear_24661702213829
// MI455X (gfx1250) — compile-verified, matrix operand order checked by reading, not run on hardware
//
#include <hip/hip_runtime.h>

// Problem dims (from reference): M = B*S, N = D_OUT, K = D_IN
#define M_DIM 8192
#define N_DIM 16384
#define K_DIM 4096

// GEMM tiling: block = 128(M) x 256(N) x 32(K), 8 waves of 32 (wave32),
// wave tile = 64x64 = 4x4 WMMA 16x16 f32 accumulators.
#define BM 128
#define BN 256
#define BK 32
#define LDA (BK + 8)   // padded LDS row stride (elements): 80B rows, 16B-aligned, bank-conflict-free b128 loads
#define LDB (BK + 8)

typedef __attribute__((ext_vector_type(16))) __bf16 v16bf;
typedef __attribute__((ext_vector_type(8)))  __bf16 v8bf;
typedef __attribute__((ext_vector_type(4)))  __bf16 v4bf;
typedef __attribute__((ext_vector_type(8)))  float  v8f;
typedef __attribute__((ext_vector_type(8)))  int    v8i;
typedef __attribute__((ext_vector_type(4)))  int    v4i;
typedef __attribute__((ext_vector_type(4)))  unsigned v4u;

// ---------------------------------------------------------------------------
// Pass 1: weights int32 (int8-range) -> bf16 (exact for |q| <= 128)
// ---------------------------------------------------------------------------
__global__ __launch_bounds__(256) void wq_to_bf16_kernel(const int* __restrict__ q,
                                                         __bf16* __restrict__ w) {
  size_t i = ((size_t)blockIdx.x * 256 + threadIdx.x) * 8;
  v8i a = *(const v8i*)(q + i);            // 32B: 2x global_load_b128
  v8bf o;
#pragma unroll
  for (int e = 0; e < 8; ++e) o[e] = (__bf16)(float)a[e];
  *(v8bf*)(w + i) = o;                     // 16B: global_store_b128
}

// ---------------------------------------------------------------------------
// Pass 2: x fp32 -> bf16, plus per-row sum (for exact zero-point term).
// One 256-thread block per row; 16 elements per thread.
// ---------------------------------------------------------------------------
__global__ __launch_bounds__(256) void x_prep_kernel(const float* __restrict__ x,
                                                     __bf16* __restrict__ xb,
                                                     float* __restrict__ rowsum) {
  __shared__ float red[256];
  const int row = blockIdx.x;
  const int tid = threadIdx.x;
  const float* xr = x + (size_t)row * K_DIM + tid * 16;
  __bf16* xbr     = xb + (size_t)row * K_DIM + tid * 16;
  float s = 0.f;
#pragma unroll
  for (int c = 0; c < 4; ++c) {
    float4 f = *(const float4*)(xr + c * 4);
    s += (f.x + f.y) + (f.z + f.w);
    v4bf o;
    o[0] = (__bf16)f.x; o[1] = (__bf16)f.y; o[2] = (__bf16)f.z; o[3] = (__bf16)f.w;
    *(v4bf*)(xbr + c * 4) = o;
  }
  red[tid] = s;
  __syncthreads();
#pragma unroll
  for (int off = 128; off > 0; off >>= 1) {
    if (tid < off) red[tid] += red[tid + off];
    __syncthreads();
  }
  if (tid == 0) rowsum[row] = red[0];
}

// ---------------------------------------------------------------------------
// TDM tile load: rows x 32 bf16 tile, row stride K_DIM elements, into LDS with
// +16B padding per 64B row (matches LDA/LDB = BK+8 layout).
// D# group0: count=1 | lds_addr | global_addr | type=2 ("image")
// D# group1: data_size=2B, pad_enable, pad_interval=3 (16 DW = 64B),
//            pad_amount=3 (4 DW = 16B), tensor_dim0=K_DIM, tensor_dim1=rows,
//            tile_dim0=BK, tile_dim1=rows, tensor_dim0_stride=K_DIM.
// ---------------------------------------------------------------------------
__device__ __forceinline__ void tdm_load_tile(unsigned lds_addr, const void* gaddr,
                                              unsigned rows) {
  unsigned long long ga = (unsigned long long)(uintptr_t)gaddr;
  v4u g0;
  g0[0] = 1u;                                        // count=1 (valid user descriptor)
  g0[1] = lds_addr;                                  // lds_addr [63:32]
  g0[2] = (unsigned)(ga & 0xffffffffu);              // global_addr [95:64]
  g0[3] = (unsigned)((ga >> 32) & 0x01ffffffu)       // global_addr [120:96]
        | (2u << 30);                                // type=2 [127:126]
  v8i g1;
  g1[0] = (int)((1u << 16)      // data_size = 2 bytes
              | (1u << 20)      // pad_enable
              | (3u << 22)      // pad_interval: 2^3 * 8B = 64B between pads
              | (3u << 25));    // pad_amount: 4 DWORDs = 16B
  g1[1] = (int)(((unsigned)K_DIM & 0xffffu) << 16);  // tensor_dim0[15:0] @ [63:48]
  g1[2] = (int)((((unsigned)K_DIM >> 16) & 0xffffu)  // tensor_dim0[31:16]
              | ((rows & 0xffffu) << 16));           // tensor_dim1[15:0]
  g1[3] = (int)(((rows >> 16) & 0xffffu)             // tensor_dim1[31:16]
              | ((unsigned)BK << 16));               // tile_dim0 @ [127:112]
  g1[4] = (int)(rows & 0xffffu);                     // tile_dim1 (tile_dim2=0)
  g1[5] = (int)K_DIM;                                // tensor_dim0_stride[31:0]
  g1[6] = 0;                                         // stride hi / dim1_stride lo
  g1[7] = 0;
  v4i z4 = {0, 0, 0, 0};                             // 2D tensor: groups 2/3 unused
  v8i z8 = {0, 0, 0, 0, 0, 0, 0, 0};
  // amdgpu-toolchain (clang-23) 6-arg form: (g0, g1, g2, g3, extra, cpol)
  __builtin_amdgcn_tensor_load_to_lds(g0, g1, z4, z4, z8, 0);
}

// ---------------------------------------------------------------------------
// Pass 3: bf16 WMMA GEMM: out[m,n] = scale[n]*(X@Qt)[m,n] + zp[n]*rowsum[m] + bias[n]
// X: M x K bf16 row-major; W: N x K bf16 row-major (== B-matrix, K-major).
// Tile staging via TENSOR_LOAD_TO_LDS (TDM DMA engine, TENSORcnt-tracked):
// wave 0 issues one descriptor per tile; zero staging VALU/VMEM in the
// other 7 waves' instruction streams.
// ---------------------------------------------------------------------------
__global__ __launch_bounds__(256) void qlinear_wmma_kernel(
    const __bf16* __restrict__ Xb, const __bf16* __restrict__ Wb,
    const float* __restrict__ rowsum, const float* __restrict__ scale,
    const float* __restrict__ zp, const float* __restrict__ bias,
    float* __restrict__ out) {
  __shared__ __bf16 As[2][BM * LDA];   // 2 x 10240 B
  __shared__ __bf16 Bs[2][BN * LDB];   // 2 x 20480 B  (total 60 KB of 320 KB/WGP)

  const int tid  = threadIdx.x;
  const int lane = tid & 31;
  const int wave = tid >> 5;       // 0..7
  const int wm   = wave & 1;       // 2 waves over M
  const int wn   = wave >> 1;      // 4 waves over N

  const size_t blockM = (size_t)blockIdx.y * BM;
  const size_t blockN = (size_t)blockIdx.x * BN;

  const char* Xp = (const char*)(Xb + blockM * K_DIM);
  const char* Wp = (const char*)(Wb + blockN * K_DIM);

  // Generic pointers to __shared__ carry the LDS byte offset in addr[31:0].
  const unsigned asBase = (unsigned)(unsigned long long)(uintptr_t)&As[0][0];
  const unsigned bsBase = (unsigned)(unsigned long long)(uintptr_t)&Bs[0][0];

  v8f acc[4][4];
#pragma unroll
  for (int i = 0; i < 4; ++i)
#pragma unroll
    for (int j = 0; j < 4; ++j)
#pragma unroll
      for (int r = 0; r < 8; ++r) acc[i][j][r] = 0.f;

  // Prologue: DMA slice k0=0 into buffer 0.
  if (wave == 0) {
    tdm_load_tile(asBase, Xp, BM);
    tdm_load_tile(bsBase, Wp, BN);
  }
  __builtin_amdgcn_s_wait_tensorcnt(0);
  __syncthreads();

  // Per-lane fragment addressing (ISA 16-bit A 16x32 layout):
  // lanes 0-15: row = lane,    K = 0..7  and 16..23
  // lanes16-31: row = lane-16, K = 8..15 and 24..31
  const int l15  = lane & 15;
  const int koff = (lane >> 4) * 8;
  const int aRow0 = wm * 64 + l15;
  const int bRow0 = wn * 64 + l15;

  int buf = 0;
  for (int k0 = 0; k0 < K_DIM; k0 += BK) {
    if ((k0 + BK < K_DIM) && wave == 0) {            // DMA next slice, overlapped with math
      unsigned nb = (unsigned)(buf ^ 1);
      tdm_load_tile(asBase + nb * (BM * LDA * 2u), Xp + (size_t)(k0 + BK) * 2, BM);
      tdm_load_tile(bsBase + nb * (BN * LDB * 2u), Wp + (size_t)(k0 + BK) * 2, BN);
    }

    v16bf af[4], bfr[4];
#pragma unroll
    for (int i = 0; i < 4; ++i) {
      const __bf16* p = &As[buf][(aRow0 + i * 16) * LDA + koff];
      v8bf lo = *(const v8bf*)p;          // ds_load_b128
      v8bf hi = *(const v8bf*)(p + 16);   // ds_load_b128
      af[i] = __builtin_shufflevector(lo, hi, 0, 1, 2, 3, 4, 5, 6, 7,
                                      8, 9, 10, 11, 12, 13, 14, 15);
    }
#pragma unroll
    for (int j = 0; j < 4; ++j) {
      const __bf16* p = &Bs[buf][(bRow0 + j * 16) * LDB + koff];
      v8bf lo = *(const v8bf*)p;
      v8bf hi = *(const v8bf*)(p + 16);
      bfr[j] = __builtin_shufflevector(lo, hi, 0, 1, 2, 3, 4, 5, 6, 7,
                                       8, 9, 10, 11, 12, 13, 14, 15);
    }

#pragma unroll
    for (int i = 0; i < 4; ++i)
#pragma unroll
      for (int j = 0; j < 4; ++j)
        acc[i][j] = __builtin_amdgcn_wmma_f32_16x16x32_bf16(
            false, af[i], false, bfr[j], (short)0, acc[i][j], false, false);

    // Drain TDM (no-op for waves 1-7), then publish the new buffer.
    __builtin_amdgcn_s_wait_tensorcnt(0);
    __syncthreads();
    buf ^= 1;
  }

  // Epilogue: out = scale[n]*acc + zp[n]*rowsum[m] + bias[n]
  // C layout: lanes 0-15: N=lane, VGPR r -> M=r; lanes 16-31: N=lane-16, M=r+8.
  const size_t outM0 = blockM + wm * 64;
  const size_t outN0 = blockN + wn * 64;
  const int mhalf = (lane >> 4) * 8;
#pragma unroll
  for (int j = 0; j < 4; ++j) {
    size_t n = outN0 + j * 16 + l15;
    float sc = scale[n], z = zp[n], bs = bias[n];
#pragma unroll
    for (int i = 0; i < 4; ++i) {
      size_t mrow = outM0 + i * 16 + mhalf;
#pragma unroll
      for (int r = 0; r < 8; ++r) {
        size_t m = mrow + r;
        out[m * N_DIM + n] = sc * acc[i][j][r] + z * rowsum[m] + bs;
      }
    }
  }
}

// ---------------------------------------------------------------------------
extern "C" void kernel_launch(void* const* d_in, const int* in_sizes, int n_in,
                              void* d_out, int out_size, void* d_ws, size_t ws_size,
                              hipStream_t stream) {
  const float* x     = (const float*)d_in[0];
  const int*   wq    = (const int*)d_in[1];
  const float* scale = (const float*)d_in[2];
  const float* zp    = (const float*)d_in[3];
  const float* bias  = (const float*)d_in[4];
  float* out = (float*)d_out;

  // Workspace layout: Wbf16 (128 MB) | Xbf16 (64 MB) | rowsum (32 KB)
  char* ws = (char*)d_ws;
  __bf16* wbf = (__bf16*)ws;
  __bf16* xbf = (__bf16*)(ws + (size_t)N_DIM * K_DIM * sizeof(__bf16));
  float* rowsum = (float*)(ws + (size_t)N_DIM * K_DIM * sizeof(__bf16)
                              + (size_t)M_DIM * K_DIM * sizeof(__bf16));

  // Pass 1: dequant-free weight narrowing (exact in bf16)
  int wblocks = (int)(((size_t)N_DIM * K_DIM) / (256 * 8));  // 32768
  wq_to_bf16_kernel<<<wblocks, 256, 0, stream>>>(wq, wbf);

  // Pass 2: activations -> bf16 + row sums
  x_prep_kernel<<<M_DIM, 256, 0, stream>>>(x, xbf, rowsum);

  // Pass 3: WMMA GEMM + affine epilogue
  dim3 grid(N_DIM / BN, M_DIM / BM);  // 64 x 64
  qlinear_wmma_kernel<<<grid, 256, 0, stream>>>(xbf, wbf, rowsum, scale, zp, bias, out);
}
